// MultiHeadAttention3D_17540646437769
// MI455X (gfx1250) — matrix operand order repaired and checked by reading, not compiled, not run on hardware
//
#include <hip/hip_runtime.h>
#include <stdint.h>

#define B_  2
#define S_  192
#define D_  512
#define H_  8
#define HD_ 64

typedef __attribute__((ext_vector_type(16))) __bf16 v16bf;
typedef __attribute__((ext_vector_type(8)))  float  v8f;

// ---------- async global->LDS copy (ASYNCcnt path), inline asm ----------
__device__ __forceinline__ void async_g2l_b128(unsigned lds_off, const void* gptr) {
  asm volatile("global_load_async_to_lds_b128 %0, %1, off"
               :: "v"(lds_off), "v"((unsigned long long)(uintptr_t)gptr)
               : "memory");
}
__device__ __forceinline__ void wait_async0() {
#if __has_builtin(__builtin_amdgcn_s_wait_asynccnt)
  __builtin_amdgcn_s_wait_asynccnt(0);
#else
  asm volatile("s_wait_asynccnt 0x0" ::: "memory");
#endif
}

// ---------- bf16 helpers (bit-level, RNE) ----------
__device__ __forceinline__ unsigned short f2bf_bits(float f) {
  union { float f; unsigned u; } v; v.f = f;
  unsigned r = v.u + 0x7FFFu + ((v.u >> 16) & 1u);
  return (unsigned short)(r >> 16);
}
__device__ __forceinline__ float bf2f(unsigned short b) {
  union { unsigned u; float f; } v; v.u = ((unsigned)b) << 16;
  return v.f;
}

// ---------- fragment-ready LDS swizzles (inverse of CDNA5 WMMA VGPR layout) ----
// A (16x32): lane = m + 16*((k>>3)&1); j = (k&7) + 8*(k>>4)
__device__ __forceinline__ int swzA(int m, int k) {
  return ((m & 15) + (((k >> 3) & 1) << 4)) * 16 + ((k & 7) + ((k >> 4) << 3));
}
// B (32x16): lane = n + 16*((k>>3)&1); j = (k&7) + 8*(k>>4)
__device__ __forceinline__ int swzB(int k, int n) {
  return ((n & 15) + (((k >> 3) & 1) << 4)) * 16 + ((k & 7) + (((k >> 4) & 1) << 3));
}
// one fragment = 32 contiguous bytes per lane -> 2x ds_load_b128, conflict-free
__device__ __forceinline__ v16bf load_frag(const unsigned short* tile, int lane) {
  return *(const v16bf*)(tile + lane * 16);
}
__device__ __forceinline__ v8f wmma_bf16(v16bf a, v16bf b, v8f c) {
  return __builtin_amdgcn_wmma_f32_16x16x32_bf16(false, a, false, b, (short)0, c,
                                                 false, false);
}

// =====================================================================
// Kernel 1: projections  y = x @ W^T  (M=384, N=512, K=512), bf16 out,
// head-split layout dst[((b*H+h)*S+s)*64+d].  blockIdx.z selects projection.
// =====================================================================
__global__ __launch_bounds__(256) void proj_kernel(
    const float* __restrict__ q, const float* __restrict__ k,
    const float* __restrict__ l, const float* __restrict__ v,
    const float* __restrict__ Wq, const float* __restrict__ Wk,
    const float* __restrict__ Wv,
    unsigned short* __restrict__ qh, unsigned short* __restrict__ kh,
    unsigned short* __restrict__ lh, unsigned short* __restrict__ vh) {
  // 8 tile-rows x (32 lanes x 16 bf16) each, fragment-ready
  __shared__ __align__(32) unsigned short As[8 * 512];
  __shared__ __align__(32) unsigned short Bs[8 * 512];
  const int tid = threadIdx.x, lane = tid & 31, wave = tid >> 5;
  const int wr = wave >> 1, wc = wave & 1;
  const int m0 = blockIdx.y * 128, n0 = blockIdx.x * 128;
  const int gz = blockIdx.z;
  const float* X = (gz == 0) ? q : (gz == 1) ? k : (gz == 2) ? l : v;
  const float* W = (gz == 0) ? Wq : (gz == 3) ? Wv : Wk;  // l uses Wk (faithful)
  unsigned short* dst = (gz == 0) ? qh : (gz == 1) ? kh : (gz == 2) ? lh : vh;

  v8f acc[2][4];
#pragma unroll
  for (int r = 0; r < 2; ++r)
#pragma unroll
    for (int c = 0; c < 4; ++c) acc[r][c] = (v8f){0,0,0,0,0,0,0,0};

  for (int kc = 0; kc < 512; kc += 32) {
    for (int i = tid; i < 128 * 32; i += 256) {
      int r = i >> 5, c = i & 31;
      As[(r >> 4) * 512 + swzA(r & 15, c)] = f2bf_bits(X[(m0 + r) * 512 + kc + c]);
    }
    for (int i = tid; i < 32 * 128; i += 256) {
      int kk = i & 31, nn = i >> 5;  // consecutive lanes -> consecutive kk (coalesced)
      Bs[(nn >> 4) * 512 + swzB(kk, nn & 15)] = f2bf_bits(W[(n0 + nn) * 512 + kc + kk]);
    }
    if (kc + 32 < 512) {  // global_prefetch_b8 path
      __builtin_prefetch(&X[(m0 + (tid >> 1)) * 512 + kc + 32], 0, 1);
      __builtin_prefetch(&W[(n0 + (tid >> 1)) * 512 + kc + 32], 0, 1);
    }
    __syncthreads();
    v16bf af[2], bfr[4];
#pragma unroll
    for (int r = 0; r < 2; ++r) af[r] = load_frag(As + (wr * 2 + r) * 512, lane);
#pragma unroll
    for (int c = 0; c < 4; ++c) bfr[c] = load_frag(Bs + (wc * 4 + c) * 512, lane);
#pragma unroll
    for (int r = 0; r < 2; ++r)
#pragma unroll
      for (int c = 0; c < 4; ++c)
        acc[r][c] = wmma_bf16(af[r], bfr[c], acc[r][c]);
    __syncthreads();
  }
  const int n_ = lane & 15;
  const int mb = (lane >> 4) << 3;
#pragma unroll
  for (int r = 0; r < 2; ++r)
#pragma unroll
    for (int c = 0; c < 4; ++c)
#pragma unroll
      for (int e = 0; e < 8; ++e) {
        int gm = m0 + wr * 32 + r * 16 + mb + e;
        int gn = n0 + wc * 64 + c * 16 + n_;
        int bb = gm / S_, s = gm % S_;
        int h = gn >> 6, d = gn & 63;
        dst[(((bb * H_ + h) * S_ + s) << 6) + d] = f2bf_bits(acc[r][c][e]);
      }
}

// =====================================================================
// Kernel 2: per-(b,h,q) fused rank-3 attention:
//   S = (K ∘ q_row/8) @ L^T  (192x64x192 bf16 WMMA -> f32 LDS)
//   joint softmax stats only (no P materialization): w[k]=sum_l exp(S-m)
//   out = (w/Z) @ V
// =====================================================================
#define LDS_ 196  // f32 leading dim for score tile (rows stay 16B aligned)

__global__ __launch_bounds__(256) void attn_kernel(
    const unsigned short* __restrict__ qh, const unsigned short* __restrict__ kh,
    const unsigned short* __restrict__ lh, const unsigned short* __restrict__ vh,
    unsigned short* __restrict__ att) {
  extern __shared__ __align__(32) char smem[];
  float* Ssm = (float*)smem;                              // 192*196 f32
  float* qs  = Ssm + 192 * LDS_;                          // 64
  float* w   = qs + 64;                                   // 192
  float* partial = w + 192;                               // 256 (+pad for 32B align)
  unsigned short* KQf = (unsigned short*)(partial + 288); // 12*2*512 bf16 (frag-ready)
  unsigned short* Lf  = KQf + 12 * 2 * 512;               // 12*2*512 bf16 (frag-ready)
  unsigned short* Vs  = Lf + 12 * 2 * 512;                // 192*64 bf16 (row-major)

  const int tid = threadIdx.x, lane = tid & 31, wave = tid >> 5;
  const int idx = blockIdx.x;
  const int qrow = idx % S_;
  const int h = (idx / S_) % H_;
  const int b = idx / (S_ * H_);
  const int headbase = ((b * H_ + h) * S_) * HD_;

  if (tid < HD_) qs[tid] = bf2f(qh[headbase + qrow * HD_ + tid]) * 0.125f;

  // V tile: pure copy -> async global->LDS (tracked by ASYNCcnt)
  {
    const unsigned vbase = (unsigned)(uintptr_t)(const void*)Vs;
    for (int i = tid; i < (S_ * HD_) / 8; i += 256)
      async_g2l_b128(vbase + i * 16, vh + headbase + i * 8);
  }
  __syncthreads();  // qs ready

  for (int i = tid; i < S_ * HD_; i += 256) {
    int kk = i >> 6, e = i & 63;
    // A = K ∘ (q/8): tile (kk>>4, e>>5), element (kk&15, e&31)
    KQf[((kk >> 4) * 2 + (e >> 5)) * 512 + swzA(kk & 15, e & 31)] =
        f2bf_bits(bf2f(kh[headbase + i]) * qs[e]);
    // B = L^T: tile col = kk>>4, n = kk&15; k-dim = e
    Lf[((kk >> 4) * 2 + (e >> 5)) * 512 + swzB(e & 31, kk & 15)] = lh[headbase + i];
  }
  wait_async0();
  __syncthreads();

  // 12x12 output tiles, 18 per wave, K=64 in two 32-steps
  for (int t = wave; t < 144; t += 8) {
    int tr = t / 12, tc = t % 12;
    v8f acc = (v8f){0,0,0,0,0,0,0,0};
#pragma unroll
    for (int ks = 0; ks < 2; ++ks) {
      v16bf a  = load_frag(KQf + (tr * 2 + ks) * 512, lane);
      v16bf bb = load_frag(Lf  + (tc * 2 + ks) * 512, lane);
      acc = wmma_bf16(a, bb, acc);
    }
    const int n_ = lane & 15, mb = (lane >> 4) << 3;
#pragma unroll
    for (int e = 0; e < 8; ++e)
      Ssm[(tr * 16 + mb + e) * LDS_ + tc * 16 + n_] = acc[e];
  }
  __syncthreads();

  // pass 1: per-row max (float4 reads, rows 16B aligned)
  float rmax = -3.0e38f;
  if (tid < S_) {
    const float* row = Ssm + tid * LDS_;
    for (int c = 0; c < S_; c += 4) {
      float4 x = *(const float4*)(row + c);
      rmax = fmaxf(rmax, fmaxf(fmaxf(x.x, x.y), fmaxf(x.z, x.w)));
    }
  }
  partial[tid] = rmax;
  __syncthreads();
  for (int s2 = 128; s2 > 0; s2 >>= 1) {
    if (tid < s2) partial[tid] = fmaxf(partial[tid], partial[tid + s2]);
    __syncthreads();
  }
  const float smax = partial[0];
  __syncthreads();

  // pass 2: fused exp + row-sum; P is never materialized
  float rsum = 0.f;
  if (tid < S_) {
    const float* row = Ssm + tid * LDS_;
    for (int c = 0; c < S_; c += 4) {
      float4 x = *(const float4*)(row + c);
      rsum += __expf(x.x - smax) + __expf(x.y - smax) +
              __expf(x.z - smax) + __expf(x.w - smax);
    }
    w[tid] = rsum;
  }
  partial[tid] = (tid < S_) ? rsum : 0.f;
  __syncthreads();
  for (int s2 = 128; s2 > 0; s2 >>= 1) {
    if (tid < s2) partial[tid] += partial[tid + s2];
    __syncthreads();
  }
  const float invZ = 1.0f / partial[0];

  // out[d] = (sum_k w[k]*V[k,d]) / Z  -> att[(b*S+q)*512 + h*64 + d]
  if (tid < HD_) {
    float acc = 0.f;
    for (int kk = 0; kk < S_; ++kk) acc += w[kk] * bf2f(Vs[kk * HD_ + tid]);
    att[(b * S_ + qrow) * D_ + h * HD_ + tid] = f2bf_bits(acc * invZ);
  }
}

// =====================================================================
// Kernel 3: output projection  out = att(384x512,bf16) @ Wout^T -> f32
// =====================================================================
__global__ __launch_bounds__(256) void out_kernel(
    const unsigned short* __restrict__ att, const float* __restrict__ Wout,
    float* __restrict__ out) {
  __shared__ __align__(32) unsigned short As[8 * 512];
  __shared__ __align__(32) unsigned short Bs[8 * 512];
  const int tid = threadIdx.x, lane = tid & 31, wave = tid >> 5;
  const int wr = wave >> 1, wc = wave & 1;
  const int m0 = blockIdx.y * 128, n0 = blockIdx.x * 128;

  v8f acc[2][4];
#pragma unroll
  for (int r = 0; r < 2; ++r)
#pragma unroll
    for (int c = 0; c < 4; ++c) acc[r][c] = (v8f){0,0,0,0,0,0,0,0};

  for (int kc = 0; kc < 512; kc += 32) {
    for (int i = tid; i < 128 * 32; i += 256) {
      int r = i >> 5, c = i & 31;
      As[(r >> 4) * 512 + swzA(r & 15, c)] = att[(m0 + r) * 512 + kc + c];
    }
    for (int i = tid; i < 32 * 128; i += 256) {
      int kk = i & 31, nn = i >> 5;
      Bs[(nn >> 4) * 512 + swzB(kk, nn & 15)] = f2bf_bits(Wout[(n0 + nn) * 512 + kc + kk]);
    }
    if (kc + 32 < 512)
      __builtin_prefetch(&Wout[(n0 + (tid >> 1)) * 512 + kc + 32], 0, 1);
    __syncthreads();
    v16bf af[2], bfr[4];
#pragma unroll
    for (int r = 0; r < 2; ++r) af[r] = load_frag(As + (wr * 2 + r) * 512, lane);
#pragma unroll
    for (int c = 0; c < 4; ++c) bfr[c] = load_frag(Bs + (wc * 4 + c) * 512, lane);
#pragma unroll
    for (int r = 0; r < 2; ++r)
#pragma unroll
      for (int c = 0; c < 4; ++c)
        acc[r][c] = wmma_bf16(af[r], bfr[c], acc[r][c]);
    __syncthreads();
  }
  const int n_ = lane & 15;
  const int mb = (lane >> 4) << 3;
#pragma unroll
  for (int r = 0; r < 2; ++r)
#pragma unroll
    for (int c = 0; c < 4; ++c)
#pragma unroll
      for (int e = 0; e < 8; ++e) {
        int gm = m0 + wr * 32 + r * 16 + mb + e;
        int gn = n0 + wc * 64 + c * 16 + n_;
        out[gm * 512 + gn] = acc[r][c][e];
      }
}

// =====================================================================
extern "C" void kernel_launch(void* const* d_in, const int* in_sizes, int n_in,
                              void* d_out, int out_size, void* d_ws, size_t ws_size,
                              hipStream_t stream) {
  const float* q    = (const float*)d_in[0];
  const float* k    = (const float*)d_in[1];
  const float* l    = (const float*)d_in[2];
  const float* v    = (const float*)d_in[3];
  const float* Wq   = (const float*)d_in[4];
  const float* Wk   = (const float*)d_in[5];
  const float* Wv   = (const float*)d_in[6];
  const float* Wout = (const float*)d_in[7];

  unsigned short* ws = (unsigned short*)d_ws;
  const size_t per = (size_t)B_ * H_ * S_ * HD_;  // 196608 bf16 elems
  unsigned short* qh  = ws;
  unsigned short* kh  = qh + per;
  unsigned short* lh  = kh + per;
  unsigned short* vh  = lh + per;
  unsigned short* att = vh + per;  // B*S*D bf16

  // 1) projections: grid (N/128=4, M/128=3, 4 projections)
  proj_kernel<<<dim3(4, 3, 4), 256, 0, stream>>>(q, k, l, v, Wq, Wk, Wv,
                                                 qh, kh, lh, vh);
  // 2) fused rank-3 attention: one block per (b,h,q)
  const size_t shb = (size_t)(192 * LDS_ + 64 + 192 + 288) * 4 +
                     (size_t)(12 * 2 * 512 * 2 + 192 * 64) * 2;
  attn_kernel<<<dim3(B_ * H_ * S_), 256, shb, stream>>>(qh, kh, lh, vh, att);
  // 3) output projection, fp32 result
  out_kernel<<<dim3(4, 3), 256, 0, stream>>>(att, Wout, (float*)d_out);
}
